// MambaLayer_56908316672059
// MI455X (gfx1250) — matrix operand order repaired and checked by reading, not compiled, not run on hardware
//
#include <hip/hip_runtime.h>

typedef __bf16 bf16;
typedef bf16 v16bf __attribute__((ext_vector_type(16)));
typedef bf16 v8bf  __attribute__((ext_vector_type(8)));
typedef float v8f  __attribute__((ext_vector_type(8)));

#define DIMC 192
#define DIL  384
#define LSEQ 16384
#define NBATCH 2
#define RTOT 32768      // NBATCH * LSEQ
#define NXDB 48         // DT_RANK(12) + 16 + 16 padded to 48
#define WH   34         // W + 2 halo columns
#define NTW  4          // N-tiles per GEMM wave
#define OBW  4          // oc-tiles per conv wave

// ---------------- helpers ----------------
__device__ __forceinline__ bf16 f2bf(float f) {
  unsigned u = __float_as_uint(f);
  unsigned r = (u + 0x7fffu + ((u >> 16) & 1u)) >> 16;   // round-to-nearest-even
  unsigned short s = (unsigned short)r;
  return __builtin_bit_cast(bf16, s);
}
__device__ __forceinline__ v16bf mk16(v8bf lo, v8bf hi) {
  return __builtin_shufflevector(lo, hi, 0,1,2,3,4,5,6,7,8,9,10,11,12,13,14,15);
}
__device__ __forceinline__ v8f wmma_bf16(v16bf a, v16bf b, v8f c) {
  return __builtin_amdgcn_wmma_f32_16x16x32_bf16(false, a, false, b, (short)0, c,
                                                 false, false);
}

// ---------------- halo-layout conversion ----------------
// halo idx = (((b*16+t)*32+h)*34 + (w+1))*192 + c
__global__ __launch_bounds__(256) void k_x_to_halo(const float* __restrict__ in,
                                                   bf16* __restrict__ out, int n) {
  int i = blockIdx.x * 256 + threadIdx.x;
  if (i >= n) return;
  int c = i % DIMC;
  int r = i / DIMC;
  int w = r & 31, h = (r >> 5) & 31, t = (r >> 10) & 15, b = r >> 14;
  out[(((size_t)((b * 16 + t) * 32 + h)) * WH + w + 1) * DIMC + c] = f2bf(in[i]);
}

__global__ __launch_bounds__(256) void k_norm_lrelu_to_halo(
    const float* __restrict__ Y, const float* __restrict__ mu,
    const float* __restrict__ rstd, bf16* __restrict__ out, int n) {
  int i = blockIdx.x * 256 + threadIdx.x;
  if (i >= n) return;
  int c = i % DIMC;
  int r = i / DIMC;
  int w = r & 31, h = (r >> 5) & 31, t = (r >> 10) & 15, b = r >> 14;
  float v = (Y[i] - mu[b * DIMC + c]) * rstd[b * DIMC + c];
  v = v >= 0.f ? v : 0.01f * v;
  out[(((size_t)((b * 16 + t) * 32 + h)) * WH + w + 1) * DIMC + c] = f2bf(v);
}

// zero the w = 0 and w = 33 apron columns (once; interiors never touch them)
__global__ __launch_bounds__(256) void k_halo_zero(bf16* __restrict__ out, int n) {
  int i = blockIdx.x * 256 + threadIdx.x;
  if (i >= n) return;                      // n = B*16*32*2*192
  int c = i % DIMC;
  int rest = i / DIMC;
  int e = rest & 1;                        // 0 -> w=0, 1 -> w=33
  int row = rest >> 1;                     // (b*16+t)*32+h
  out[((size_t)row * WH + (e ? 33 : 0)) * DIMC + c] =
      __builtin_bit_cast(bf16, (unsigned short)0);
}

// ---------------- weight packing into per-lane WMMA B layout ----------------
__global__ __launch_bounds__(256) void k_pack_dense(const float* __restrict__ W,
                                                    int Nvalid, int K,
                                                    bf16* __restrict__ dst, int NT) {
  int KC = K >> 5;
  int total = NT * KC * 512;
  int idx = blockIdx.x * 256 + threadIdx.x;
  if (idx >= total) return;
  int e = idx & 15;
  int rest = idx >> 4;
  int lane = rest & 31;
  rest >>= 5;
  int kc = rest % KC;
  int nt = rest / KC;
  int n = nt * 16 + (lane & 15);
  int k = kc * 32 + (lane >> 4) * 8 + (e & 7) + (e >> 3) * 16;
  float v = (n < Nvalid) ? W[(size_t)n * K + k] : 0.0f;
  dst[idx] = f2bf(v);
}

// conv weights (O=192,I=192,3,3,3): dst[(((tap*6+kc)*12+ob)*32+lane)*16+e]
__global__ __launch_bounds__(256) void k_pack_conv(const float* __restrict__ W,
                                                   bf16* __restrict__ dst) {
  int idx = blockIdx.x * 256 + threadIdx.x;
  if (idx >= 27 * 6 * 12 * 512) return;
  int e = idx & 15;
  int rest = idx >> 4;
  int lane = rest & 31;
  rest >>= 5;
  int ob = rest % 12;
  rest /= 12;
  int kc = rest % 6;
  int tap = rest / 6;
  int oc = ob * 16 + (lane & 15);
  int ic = kc * 32 + (lane >> 4) * 8 + (e & 7) + (e >> 3) * 16;
  dst[idx] = f2bf(W[((size_t)(oc * DIMC + ic)) * 27 + tap]);
}

// ---------------- implicit-GEMM 3D conv, halo input, 4 oc-tiles per wave -----
__global__ __launch_bounds__(32) void k_conv3d_wmma(const bf16* __restrict__ Xh,
                                                    const bf16* __restrict__ Wp,
                                                    const float* __restrict__ bias,
                                                    float* __restrict__ Y) {
  int tile = blockIdx.x;            // ((b*16+t)*32+h)*2 + wblk
  int obg = blockIdx.y;             // 0..2 -> oc tiles obg*4 .. obg*4+3
  int lane = threadIdx.x;
  int wblk = tile & 1;
  int h = (tile >> 1) & 31;
  int t = (tile >> 6) & 15;
  int b = tile >> 10;
  int m = lane & 15, g = lane >> 4;
  v8f acc[OBW];
#pragma unroll
  for (int j = 0; j < OBW; ++j)
#pragma unroll
    for (int i = 0; i < 8; ++i) acc[j][i] = 0.0f;

  for (int dz = 0; dz < 3; ++dz) {
    int tz = t + dz - 1;
    if ((unsigned)tz >= 16u) continue;
    for (int dy = 0; dy < 3; ++dy) {
      int hy = h + dy - 1;
      if ((unsigned)hy >= 32u) continue;
#pragma unroll
      for (int dx = 0; dx < 3; ++dx) {
        int tap = (dz * 3 + dy) * 3 + dx;
        // halo: no bounds check needed, apron columns are zero
        const bf16* ab =
            Xh + (((size_t)((b * 16 + tz) * 32 + hy)) * WH + wblk * 16 + m + dx) * DIMC +
            g * 8;
        const bf16* bt = Wp + (size_t)(tap * 6 * 12 + obg * OBW) * 512 + lane * 16;
#pragma unroll
        for (int kc = 0; kc < 6; ++kc) {
          v8bf lo = *(const v8bf*)(ab + kc * 32);
          v8bf hi = *(const v8bf*)(ab + kc * 32 + 16);
          v16bf av = mk16(lo, hi);
          // Load all 4 B fragments into distinct live registers, then issue
          // 4 back-to-back WMMAs. The empty asm pins all four values live
          // simultaneously so RA cannot coalesce them into one tuple (which
          // would re-serialize into load->wait->wmma chains).
          v16bf b0 = *(const v16bf*)(bt + (size_t)(kc * 12 + 0) * 512);
          v16bf b1 = *(const v16bf*)(bt + (size_t)(kc * 12 + 1) * 512);
          v16bf b2 = *(const v16bf*)(bt + (size_t)(kc * 12 + 2) * 512);
          v16bf b3 = *(const v16bf*)(bt + (size_t)(kc * 12 + 3) * 512);
          asm volatile("" : "+v"(b0), "+v"(b1), "+v"(b2), "+v"(b3));
          acc[0] = wmma_bf16(av, b0, acc[0]);
          acc[1] = wmma_bf16(av, b1, acc[1]);
          acc[2] = wmma_bf16(av, b2, acc[2]);
          acc[3] = wmma_bf16(av, b3, acc[3]);
        }
      }
    }
  }
  int posbase = (t * 32 + h) * 32 + wblk * 16;
#pragma unroll
  for (int j = 0; j < OBW; ++j) {
    int ob = obg * OBW + j;
    float bb = bias[ob * 16 + m];
    float* yb = Y + ((size_t)b * LSEQ + posbase + g * 8) * DIMC + ob * 16 + m;
#pragma unroll
    for (int i = 0; i < 8; ++i) yb[(size_t)i * DIMC] = acc[j][i] + bb;
  }
}

// ---------------- instance-norm statistics ----------------
__global__ __launch_bounds__(256) void k_inorm_stats(const float* __restrict__ Y,
                                                     float* __restrict__ mu,
                                                     float* __restrict__ rstd) {
  int bc = blockIdx.x;
  int c = bc % DIMC, b = bc / DIMC;
  const float* p = Y + (size_t)b * LSEQ * DIMC + c;
  float s = 0.f, s2 = 0.f;
  for (int i = threadIdx.x; i < LSEQ; i += 256) {
    float v = p[(size_t)i * DIMC];
    s += v;
    s2 += v * v;
  }
  __shared__ float sh[256], sh2[256];
  sh[threadIdx.x] = s;
  sh2[threadIdx.x] = s2;
  __syncthreads();
  for (int o = 128; o; o >>= 1) {
    if (threadIdx.x < o) {
      sh[threadIdx.x] += sh[threadIdx.x + o];
      sh2[threadIdx.x] += sh2[threadIdx.x + o];
    }
    __syncthreads();
  }
  if (threadIdx.x == 0) {
    float m = sh[0] * (1.0f / LSEQ);
    float var = sh2[0] * (1.0f / LSEQ) - m * m;
    mu[bc] = m;
    rstd[bc] = rsqrtf(var + 1e-5f);
  }
}

__global__ __launch_bounds__(256) void k_norm_resid(const float* __restrict__ Y,
                                                    const float* __restrict__ mu,
                                                    const float* __restrict__ rstd,
                                                    const float* __restrict__ Xin,
                                                    float* __restrict__ out, int n) {
  int i = blockIdx.x * 256 + threadIdx.x;
  if (i >= n) return;
  int c = i % DIMC;
  int b = i / (LSEQ * DIMC);
  out[i] = (Y[i] - mu[b * DIMC + c]) * rstd[b * DIMC + c] + Xin[i];
}

// ---------------- LayerNorm on the reinterpreted buffer ----------------
__global__ __launch_bounds__(32) void k_layernorm(const float* __restrict__ XCF,
                                                  const float* __restrict__ lng,
                                                  const float* __restrict__ lnb,
                                                  bf16* __restrict__ out) {
  int bl = blockIdx.x;
  int b = bl >> 14;
  int l = bl & (LSEQ - 1);
  const float* base = XCF + (size_t)b * LSEQ * DIMC + l;
  float v[6];
  float s = 0.f;
#pragma unroll
  for (int i = 0; i < 6; ++i) {
    v[i] = base[(size_t)(threadIdx.x + 32 * i) * LSEQ];
    s += v[i];
  }
  for (int o = 16; o; o >>= 1) s += __shfl_xor(s, o, 32);
  float mean = s * (1.0f / DIMC);
  float s2 = 0.f;
#pragma unroll
  for (int i = 0; i < 6; ++i) {
    float d = v[i] - mean;
    s2 += d * d;
  }
  for (int o = 16; o; o >>= 1) s2 += __shfl_xor(s2, o, 32);
  float rs = rsqrtf(s2 * (1.0f / DIMC) + 1e-5f);
  bf16* ob = out + (size_t)bl * DIMC;
#pragma unroll
  for (int i = 0; i < 6; ++i) {
    int d = threadIdx.x + 32 * i;
    ob[d] = f2bf((v[i] - mean) * rs * lng[d] + lnb[d]);
  }
}

// ------------- dense WMMA GEMM: 4 waves/block, B staged in LDS ---------------
// block: 4 waves = 4 consecutive M-tiles; blockIdx.y selects a group of 4 N-tiles
__global__ __launch_bounds__(128) void k_gemm_bf16(const bf16* __restrict__ A, int K,
                                                   const bf16* __restrict__ Bpk,
                                                   float* __restrict__ C, int ldc,
                                                   int Nactive) {
  __shared__ __align__(64) bf16 sB[NTW * 12 * 512];   // up to KC=12 -> 48KB
  int KC = K >> 5;
  int tid = threadIdx.x;
  int wave = tid >> 5;
  int lane = tid & 31;

  {  // cooperative stage of the 4 B tiles (contiguous in packed layout)
    const bf16* src = Bpk + (size_t)blockIdx.y * NTW * KC * 512;
    int total = NTW * KC * 512;
    for (int i = tid * 16; i < total; i += 128 * 16)
      *(v16bf*)(sB + i) = *(const v16bf*)(src + i);
  }
  __syncthreads();

  int mt = blockIdx.x * 4 + wave;
  int rbase = mt * 16;
  int m = lane & 15, g = lane >> 4;
  const bf16* arow = A + (size_t)(rbase + m) * K + g * 8;
  v8f acc[NTW];
#pragma unroll
  for (int j = 0; j < NTW; ++j)
#pragma unroll
    for (int i = 0; i < 8; ++i) acc[j][i] = 0.0f;

#pragma unroll 4
  for (int kc = 0; kc < KC; ++kc) {
    __builtin_prefetch(arow + (kc + 2) * 32, 0, 1);
    v8bf lo = *(const v8bf*)(arow + kc * 32);
    v8bf hi = *(const v8bf*)(arow + kc * 32 + 16);
    v16bf av = mk16(lo, hi);
    // preload 4 B fragments from LDS into distinct live registers (see conv)
    const bf16* bp = sB + (size_t)kc * 512 + lane * 16;
    v16bf b0 = *(const v16bf*)(bp + (size_t)(0 * KC) * 512);
    v16bf b1 = *(const v16bf*)(bp + (size_t)(1 * KC) * 512);
    v16bf b2 = *(const v16bf*)(bp + (size_t)(2 * KC) * 512);
    v16bf b3 = *(const v16bf*)(bp + (size_t)(3 * KC) * 512);
    asm volatile("" : "+v"(b0), "+v"(b1), "+v"(b2), "+v"(b3));
    acc[0] = wmma_bf16(av, b0, acc[0]);
    acc[1] = wmma_bf16(av, b1, acc[1]);
    acc[2] = wmma_bf16(av, b2, acc[2]);
    acc[3] = wmma_bf16(av, b3, acc[3]);
  }
#pragma unroll
  for (int j = 0; j < NTW; ++j) {
    int ncol = (blockIdx.y * NTW + j) * 16;
    if (ncol < Nactive) {
      float* crow = C + (size_t)(rbase + g * 8) * ldc + ncol + m;
#pragma unroll
      for (int i = 0; i < 8; ++i) crow[(size_t)i * ldc] = acc[j][i];
    }
  }
}

// ---------------- depthwise causal conv1d + SiLU ----------------
__global__ __launch_bounds__(256) void k_dwconv_silu(const float* __restrict__ U,
                                                     const float* __restrict__ W,
                                                     const float* __restrict__ Bc,
                                                     float* __restrict__ US,
                                                     bf16* __restrict__ USbf, int n) {
  int idx = blockIdx.x * 256 + threadIdx.x;
  if (idx >= n) return;
  int c = idx % DIL;
  int r = idx / DIL;
  int l = r & (LSEQ - 1);
  int b = r >> 14;
  float acc = Bc[c];
#pragma unroll
  for (int j = 0; j < 4; ++j) {
    int ls = l - 3 + j;
    if (ls >= 0) acc += W[c * 4 + j] * U[((size_t)(b << 14) + ls) * DIL + c];
  }
  float sv = acc / (1.0f + __expf(-acc));
  US[idx] = sv;
  USbf[idx] = f2bf(sv);
}

// ---------------- delta = softplus(dt @ dt_proj^T + b) ----------------
__global__ __launch_bounds__(256) void k_delta(const float* __restrict__ XDB,
                                               const float* __restrict__ DTW,
                                               const float* __restrict__ DTB,
                                               float* __restrict__ DELTA, int n) {
  int idx = blockIdx.x * 256 + threadIdx.x;
  if (idx >= n) return;
  int d = idx % DIL;
  int r = idx / DIL;
  const float* dtp = XDB + (size_t)r * NXDB;
  float acc = DTB[d];
#pragma unroll
  for (int j = 0; j < 12; ++j) acc += dtp[j] * DTW[d * 12 + j];
  DELTA[idx] = (acc > 20.f) ? acc : log1pf(__expf(acc));
}

__global__ __launch_bounds__(256) void k_prepA(const float* __restrict__ Alog,
                                               float* __restrict__ A, int n) {
  int i = blockIdx.x * 256 + threadIdx.x;
  if (i < n) A[i] = -__expf(Alog[i]);
}

// ---------------- chunked selective scan (64 chunks x 256 steps) -------------
#define NCHUNK 64
#define CLEN 256
__global__ __launch_bounds__(256) void k_scan1(const float* __restrict__ DELTA,
                                               const float* __restrict__ US,
                                               const float* __restrict__ XDB,
                                               const float* __restrict__ A,
                                               float* __restrict__ APROD,
                                               float* __restrict__ HLOC) {
  int gid = blockIdx.x * 256 + threadIdx.x;
  int s = gid & 15;
  int hw = gid >> 4;
  int c = hw & 63;
  int t1 = hw >> 6;
  int d = t1 % DIL;
  int b = t1 / DIL;
  float a = A[d * 16 + s];
  float ap = 1.0f, h = 0.0f;
  size_t rbase = ((size_t)b << 14) + (c << 8);
  for (int i = 0; i < CLEN; ++i) {
    size_t r = rbase + i;
    float dlt = DELTA[r * DIL + d];
    float u = US[r * DIL + d];
    float bt = XDB[r * NXDB + 12 + s];
    float e = __expf(dlt * a);
    ap *= e;
    h = e * h + dlt * u * bt;
  }
  APROD[gid] = ap;
  HLOC[gid] = h;
}

__global__ __launch_bounds__(256) void k_scan2(const float* __restrict__ APROD,
                                               const float* __restrict__ HLOC,
                                               float* __restrict__ H0, int n) {
  int idx = blockIdx.x * 256 + threadIdx.x;
  if (idx >= n) return;
  int s = idx & 15;
  int t1 = idx >> 4;
  int d = t1 % DIL;
  int b = t1 / DIL;
  size_t base = ((size_t)(b * DIL + d)) * NCHUNK * 16 + s;
  float h = 0.0f;
  for (int c = 0; c < NCHUNK; ++c) {
    size_t o = base + (size_t)c * 16;
    H0[o] = h;
    h = APROD[o] * h + HLOC[o];
  }
}

__global__ __launch_bounds__(256) void k_scan3(const float* __restrict__ DELTA,
                                               const float* __restrict__ US,
                                               const float* __restrict__ XDB,
                                               const float* __restrict__ A,
                                               const float* __restrict__ H0,
                                               float* __restrict__ YS) {
  int gid = blockIdx.x * 256 + threadIdx.x;
  int s = gid & 15;
  int hw = gid >> 4;
  int c = hw & 63;
  int t1 = hw >> 6;
  int d = t1 % DIL;
  int b = t1 / DIL;
  float a = A[d * 16 + s];
  float h = H0[gid];
  size_t rbase = ((size_t)b << 14) + (c << 8);
  for (int i = 0; i < CLEN; ++i) {
    size_t r = rbase + i;
    float dlt = DELTA[r * DIL + d];
    float u = US[r * DIL + d];
    float bt = XDB[r * NXDB + 12 + s];
    float ct = XDB[r * NXDB + 28 + s];
    float e = __expf(dlt * a);
    h = e * h + dlt * u * bt;
    float p = h * ct;
    p += __shfl_xor(p, 8, 32);
    p += __shfl_xor(p, 4, 32);
    p += __shfl_xor(p, 2, 32);
    p += __shfl_xor(p, 1, 32);
    if (s == 0) YS[r * DIL + d] = p;
  }
}

// ---------------- y = (ys + u*Dp) * silu(z), to bf16 ----------------
__global__ __launch_bounds__(256) void k_ypost(const float* __restrict__ YS,
                                               const float* __restrict__ US,
                                               const float* __restrict__ Z,
                                               const float* __restrict__ Dp,
                                               bf16* __restrict__ YBF, int n) {
  int idx = blockIdx.x * 256 + threadIdx.x;
  if (idx >= n) return;
  int c = idx % DIL;
  float y = YS[idx] + US[idx] * Dp[c];
  float z = Z[idx];
  y *= z / (1.0f + __expf(-z));
  YBF[idx] = f2bf(y);
}

// =============================== host side ===================================
extern "C" void kernel_launch(void* const* d_in, const int* in_sizes, int n_in,
                              void* d_out, int out_size, void* d_ws, size_t ws_size,
                              hipStream_t stream) {
  const float* x    = (const float*)d_in[0];
  const float* c1w  = (const float*)d_in[1];
  const float* c1b  = (const float*)d_in[2];
  const float* c2w  = (const float*)d_in[3];
  const float* c2b  = (const float*)d_in[4];
  const float* lng  = (const float*)d_in[5];
  const float* lnb  = (const float*)d_in[6];
  const float* ipw  = (const float*)d_in[7];
  const float* c1dw = (const float*)d_in[8];
  const float* c1db = (const float*)d_in[9];
  const float* xpw  = (const float*)d_in[10];
  const float* dtw  = (const float*)d_in[11];
  const float* dtb  = (const float*)d_in[12];
  const float* alog = (const float*)d_in[13];
  const float* Dp   = (const float*)d_in[14];
  const float* opw  = (const float*)d_in[15];

  char* ws = (char*)d_ws;
  size_t off = 0;
  auto alloc = [&](size_t bytes) -> void* {
    off = (off + 255) & ~(size_t)255;
    void* p = ws + off;
    off += bytes;
    return p;
  };

  const size_t NE_C = (size_t)RTOT * DIMC;              // 6.29M
  const size_t NE_D = (size_t)RTOT * DIL;               // 12.58M
  const size_t NE_H = (size_t)NBATCH * 16 * 32 * WH * DIMC;  // halo bf16 elems
  const size_t NHEDGE = (size_t)NBATCH * 16 * 32 * 2 * DIMC;
  const size_t NSCAN = (size_t)NBATCH * DIL * NCHUNK * 16;   // 786432

  bf16*  wp1   = (bf16*)alloc(27 * 6 * 12 * 512 * sizeof(bf16));
  bf16*  wp2   = (bf16*)alloc(27 * 6 * 12 * 512 * sizeof(bf16));
  bf16*  wpIn  = (bf16*)alloc(48 * 6 * 512 * sizeof(bf16));
  bf16*  wpXp  = (bf16*)alloc(4 * 12 * 512 * sizeof(bf16));
  bf16*  wpOut = (bf16*)alloc(12 * 12 * 512 * sizeof(bf16));
  bf16*  r16a  = (bf16*)alloc(NE_H * sizeof(bf16));     // xh -> a1h -> xnbf
  float* r32a  = (float*)alloc(NE_C * sizeof(float));   // y1 -> y2
  float* xcf   = (float*)alloc(NE_C * sizeof(float));
  float* bufU  = (float*)alloc(NE_D * sizeof(float));   // u -> delta
  float* bufZ  = (float*)alloc(NE_D * sizeof(float));
  float* bufUS = (float*)alloc(NE_D * sizeof(float));
  bf16*  r16b  = (bf16*)alloc(NE_D * sizeof(bf16));     // usbf -> ybf
  float* xdb   = (float*)alloc((size_t)RTOT * NXDB * sizeof(float));
  float* ys    = (float*)alloc(NE_D * sizeof(float));
  float* aprod = (float*)alloc(NSCAN * sizeof(float));
  float* hloc  = (float*)alloc(NSCAN * sizeof(float));
  float* h0    = (float*)alloc(NSCAN * sizeof(float));
  float* mu1   = (float*)alloc(384 * sizeof(float));
  float* rstd1 = (float*)alloc(384 * sizeof(float));
  float* mu2   = (float*)alloc(384 * sizeof(float));
  float* rstd2 = (float*)alloc(384 * sizeof(float));
  float* Amat  = (float*)alloc(DIL * 16 * sizeof(float));
  (void)ws_size; (void)in_sizes; (void)n_in; (void)out_size;

  bf16* xh = r16a;        // phase aliases
  bf16* a1h = r16a;
  bf16* xnbf = r16a;
  float* y1 = r32a;
  float* y2 = r32a;
  float* delta = bufU;
  bf16* usbf = r16b;
  bf16* ybf = r16b;

  auto blks = [](size_t n) { return (unsigned)((n + 255) / 256); };

  // --- weight prep + halo input ---
  k_pack_conv<<<blks(27 * 6 * 12 * 512), 256, 0, stream>>>(c1w, wp1);
  k_pack_conv<<<blks(27 * 6 * 12 * 512), 256, 0, stream>>>(c2w, wp2);
  k_pack_dense<<<blks(48 * 6 * 512), 256, 0, stream>>>(ipw, 768, 192, wpIn, 48);
  k_pack_dense<<<blks(4 * 12 * 512), 256, 0, stream>>>(xpw, 44, 384, wpXp, 4);
  k_pack_dense<<<blks(12 * 12 * 512), 256, 0, stream>>>(opw, 192, 384, wpOut, 12);
  k_prepA<<<blks(DIL * 16), 256, 0, stream>>>(alog, Amat, DIL * 16);
  k_halo_zero<<<blks(NHEDGE), 256, 0, stream>>>(r16a, (int)NHEDGE);
  k_x_to_halo<<<blks(NE_C), 256, 0, stream>>>(x, xh, (int)NE_C);

  // --- conv1 -> inorm -> leaky-relu (back into halo layout) ---
  k_conv3d_wmma<<<dim3(2048, 3), 32, 0, stream>>>(xh, wp1, c1b, y1);
  k_inorm_stats<<<NBATCH * DIMC, 256, 0, stream>>>(y1, mu1, rstd1);
  k_norm_lrelu_to_halo<<<blks(NE_C), 256, 0, stream>>>(y1, mu1, rstd1, a1h, (int)NE_C);

  // --- conv2 -> inorm -> +residual ---
  k_conv3d_wmma<<<dim3(2048, 3), 32, 0, stream>>>(a1h, wp2, c2b, y2);
  k_inorm_stats<<<NBATCH * DIMC, 256, 0, stream>>>(y2, mu2, rstd2);
  k_norm_resid<<<blks(NE_C), 256, 0, stream>>>(y2, mu2, rstd2, x, xcf, (int)NE_C);

  // --- layernorm over reinterpreted buffer ---
  k_layernorm<<<RTOT, 32, 0, stream>>>(xcf, lng, lnb, xnbf);

  // --- in_proj (u | z): M=32768, K=192, N=384 each half ---
  k_gemm_bf16<<<dim3(RTOT / 64, 6), 128, 0, stream>>>(xnbf, 192, wpIn, bufU, DIL, DIL);
  k_gemm_bf16<<<dim3(RTOT / 64, 6), 128, 0, stream>>>(xnbf, 192, wpIn + 24 * 6 * 512,
                                                      bufZ, DIL, DIL);

  // --- depthwise causal conv + silu ---
  k_dwconv_silu<<<blks(NE_D), 256, 0, stream>>>(bufU, c1dw, c1db, bufUS, usbf,
                                                (int)NE_D);

  // --- x_proj (dt | B | C), N padded 44 -> 64, stores guarded at 48 ---
  k_gemm_bf16<<<dim3(RTOT / 64, 1), 128, 0, stream>>>(usbf, 384, wpXp, xdb, NXDB,
                                                      NXDB);

  // --- delta (overwrites bufU, now dead) ---
  k_delta<<<blks(NE_D), 256, 0, stream>>>(xdb, dtw, dtb, delta, (int)NE_D);

  // --- chunked selective scan ---
  k_scan1<<<blks(NSCAN), 256, 0, stream>>>(delta, bufUS, xdb, Amat, aprod, hloc);
  k_scan2<<<blks(NBATCH * DIL * 16), 256, 0, stream>>>(aprod, hloc, h0,
                                                       NBATCH * DIL * 16);
  k_scan3<<<blks(NSCAN), 256, 0, stream>>>(delta, bufUS, xdb, Amat, h0, ys);

  // --- gate + out_proj ---
  k_ypost<<<blks(NE_D), 256, 0, stream>>>(ys, bufUS, bufZ, Dp, ybf, (int)NE_D);
  k_gemm_bf16<<<dim3(RTOT / 64, 3), 128, 0, stream>>>(ybf, 384, wpOut, (float*)d_out,
                                                      DIMC, DIMC);
}
